// AttentionDevignModel_70789650972770
// MI455X (gfx1250) — matrix operand order repaired and checked by reading, not compiled, not run on hardware
//
#include <hip/hip_runtime.h>
#include <hip/hip_bf16.h>

typedef __attribute__((ext_vector_type(16))) __bf16 v16bf;
typedef __attribute__((ext_vector_type(8)))  float  v8f;
typedef unsigned short u16;

#define N_NODES  50000
#define N_EDGES  800000
#define N_GRAPHS 128
#define IN_DIM   100
#define HID      256
#define STEPS    5
#define KPAD_IN  128
#define CHUNK    8000   // multiple of 16; 50000 = 6*8000 + 2000

__device__ __forceinline__ u16 f2bf(float f) {
  unsigned u = __float_as_uint(f);
  u += 0x7fffu + ((u >> 16) & 1u);     // round-to-nearest-even
  return (u16)(u >> 16);
}

union BF16Frag { v16bf v; u16 u[16]; };

// ---------------------------------------------------------------------------
// Pack a fp32 weight matrix into bf16 WMMA B-fragment layout.
// Logical B[k][n], k<K (padded, zero-fill k>=Ksrc), n<N.
//   trans==0 : B[k][n] = W[k*ld + n]
//   trans==1 : B[k][n] = W[n*ld + k]
// Fragment layout: out[((kt*(N/16)+nt)*32 + lane)*16 + e]
//   n = nt*16 + (lane&15),  k = kt*32 + (lane>>4)*16 + e
// ---------------------------------------------------------------------------
__global__ void pack_b(const float* __restrict__ W, int ld, int trans,
                       int K, int N, u16* __restrict__ out, int Ksrc) {
  int idx = blockIdx.x * blockDim.x + threadIdx.x;
  if (idx >= K * N) return;
  int e    = idx & 15;
  int lane = (idx >> 4) & 31;
  int frag = idx >> 9;
  int ntilesTot = N >> 4;
  int kt = frag / ntilesTot;
  int nt = frag % ntilesTot;
  int n = nt * 16 + (lane & 15);
  int k = kt * 32 + (lane >> 4) * 16 + e;
  float v = 0.0f;
  if (k < Ksrc) v = trans ? W[(size_t)n * ld + k] : W[(size_t)k * ld + n];
  out[idx] = f2bf(v);
}

// ---------------------------------------------------------------------------
// C[M,N] = act(A[M,K](fp32, row-major) @ Bpack(bf16 fragments) + bias)
// One wave computes a 16 x (16*NT) tile. M%16==0, N%(16*NT)==0, K%64==0
// (ktiles even). 2-deep B double-buffer + converted-A fragment pipeline.
// ---------------------------------------------------------------------------
template<int NT, int ACT>
__global__ void __launch_bounds__(256, 1)
gemm_wmma(const float* __restrict__ A,
          const u16* __restrict__ Bp,
          const float* __restrict__ bias,
          float* __restrict__ C,
          int M, int K, int N) {
  const int lane = threadIdx.x & 31;
  const int wid  = (blockIdx.x * blockDim.x + threadIdx.x) >> 5;
  const int ntilesTot = N >> 4;
  const int nstrips   = ntilesTot / NT;
  const int mtiles    = M >> 4;
  if (wid >= mtiles * nstrips) return;           // wave-uniform exit
  const int mt  = wid / nstrips;
  const int nt0 = (wid % nstrips) * NT;

  v8f acc[NT];
#pragma unroll
  for (int t = 0; t < NT; ++t)
#pragma unroll
    for (int i = 0; i < 8; ++i) acc[t][i] = 0.0f;

  // A fragment: lanes 0-15 -> row M=lane,    K = {0..15}
  //             lanes16-31 -> row M=lane-16, K = {16..31}
  const int row   = mt * 16 + (lane & 15);
  const int khalf = (lane >> 4) * 16;
  const float* arow = A + (size_t)row * K + khalf;
  const u16* bbase  = Bp + (size_t)nt0 * 32 * 16 + (size_t)lane * 16;
  const size_t bkstride = (size_t)ntilesTot * 32 * 16;   // elems per k-tile

  float araw[16];
  BF16Frag a0, a1;
  v16bf b0[NT], b1[NT];

#define LOAD_A(kt)                                                          \
  {                                                                         \
    const float* ap_ = arow + (kt) * 32;                                    \
    _Pragma("unroll") for (int e = 0; e < 8; ++e) araw[e] = ap_[e];         \
    _Pragma("unroll") for (int e = 0; e < 8; ++e) araw[8 + e] = ap_[8 + e]; \
  }
#define CVT_A(dst)                                                          \
  {                                                                         \
    _Pragma("unroll") for (int e = 0; e < 16; ++e) dst.u[e] = f2bf(araw[e]);\
  }
#define LOAD_B(kt, dst)                                                     \
  {                                                                         \
    const u16* bp_ = bbase + (size_t)(kt) * bkstride;                       \
    _Pragma("unroll") for (int t = 0; t < NT; ++t)                          \
      dst[t] = *(const v16bf*)(bp_ + (size_t)t * (32 * 16));                \
  }
#define DO_WMMA(afr, bfr)                                                   \
  {                                                                         \
    _Pragma("unroll") for (int t = 0; t < NT; ++t)                          \
      acc[t] = __builtin_amdgcn_wmma_f32_16x16x32_bf16(                     \
          false, afr.v, false, bfr[t], (short)0, acc[t], false, false);     \
  }

  const int ktiles = K >> 5;   // even for all call sites
  LOAD_A(0);
  LOAD_B(0, b0);
  CVT_A(a0);
  LOAD_A(1);
  LOAD_B(1, b1);
  for (int kt = 0; kt < ktiles; kt += 2) {
    DO_WMMA(a0, b0);                 // consume even ktile
    CVT_A(a1);                       // finalize odd ktile's A fragment
    const bool more = (kt + 2 < ktiles);
    if (more) {
      LOAD_A(kt + 2);
      LOAD_B(kt + 2, b0);
    }
    DO_WMMA(a1, b1);                 // consume odd ktile
    if (more) {
      CVT_A(a0);                     // next even A fragment
      LOAD_A(kt + 3);                // next odd raw A (kt+3 <= ktiles-1)
      LOAD_B(kt + 3, b1);
    }
  }
#undef LOAD_A
#undef CVT_A
#undef LOAD_B
#undef DO_WMMA

  // C/D layout: lane n = lane&15; VGPR i -> row i + 8*(lane>=16)
  const int colb   = lane & 15;
  const int rowoff = (lane >> 4) * 8;
#pragma unroll
  for (int t = 0; t < NT; ++t) {
    const int col = (nt0 + t) * 16 + colb;
    const float bv = bias ? bias[col] : 0.0f;
#pragma unroll
    for (int i = 0; i < 8; ++i) {
      float v = acc[t][i] + bv;
      if (ACT == 1) v = fmaxf(v, 0.0f);
      C[(size_t)(mt * 16 + rowoff + i) * N + col] = v;
    }
  }
}

// ---------------------------------------------------------------------------
__global__ void fill_f32(float* __restrict__ p, float v, long long n) {
  long long t = (long long)blockIdx.x * blockDim.x + threadIdx.x;
  if (t < n) p[t] = v;
}

__global__ void pad_x(const float* __restrict__ x, float* __restrict__ xp) {
  int idx = blockIdx.x * blockDim.x + threadIdx.x;
  if (idx >= N_NODES * KPAD_IN) return;
  int n = idx >> 7, c = idx & 127;
  xp[idx] = (c < IN_DIM) ? x[(size_t)n * IN_DIM + c] : 0.0f;
}

// agg[dst[e]] += m[src[e]] ; 64 float4 chunks (=256 floats) per edge
__global__ void scatter_add(const float* __restrict__ m, const int* __restrict__ src,
                            const int* __restrict__ dst, float* __restrict__ agg) {
  long long t = (long long)blockIdx.x * blockDim.x + threadIdx.x;
  int e = (int)(t >> 6);
  if (e >= N_EDGES) return;
  int c = (int)(t & 63) << 2;
  const float4 v = *(const float4*)(m + (size_t)src[e] * HID + c);
  float* o = agg + (size_t)dst[e] * HID + c;
  unsafeAtomicAdd(o + 0, v.x);
  unsafeAtomicAdd(o + 1, v.y);
  unsafeAtomicAdd(o + 2, v.z);
  unsafeAtomicAdd(o + 3, v.w);
}

// GRU cell, in-place update of h rows [base, base+cm)
__global__ void gru_cell(const float* __restrict__ gi, const float* __restrict__ gh,
                         float* __restrict__ h, int base, int cm) {
  int t = blockIdx.x * blockDim.x + threadIdx.x;
  if (t >= cm * HID) return;
  int n = t >> 8, c = t & 255;
  const float* gir = gi + (size_t)n * (3 * HID);
  const float* ghr = gh + (size_t)n * (3 * HID);
  float i_r = gir[c], i_z = gir[HID + c], i_n = gir[2 * HID + c];
  float h_r = ghr[c], h_z = ghr[HID + c], h_n = ghr[2 * HID + c];
  float r  = 1.0f / (1.0f + __expf(-(i_r + h_r)));
  float z  = 1.0f / (1.0f + __expf(-(i_z + h_z)));
  float nn = tanhf(i_n + r * h_n);
  size_t hi = (size_t)(base + n) * HID + c;
  float hv = h[hi];
  h[hi] = (1.0f - z) * nn + z * hv;
}

__global__ void relu_ip(float* __restrict__ p, long long n) {
  long long t = (long long)blockIdx.x * blockDim.x + threadIdx.x;
  if (t < n) p[t] = fmaxf(p[t], 0.0f);
}

// a[n] = sigmoid(dot(a1[n,:128], Wa2) + ba2)
__global__ void attention(const float* __restrict__ a1, const float* __restrict__ Wa2,
                          const float* __restrict__ ba2, float* __restrict__ aout) {
  int n = blockIdx.x * blockDim.x + threadIdx.x;
  if (n >= N_NODES) return;
  const float* r = a1 + (size_t)n * (HID / 2);
  float s = ba2[0];
#pragma unroll 8
  for (int k = 0; k < HID / 2; ++k) s += r[k] * Wa2[k];
  aout[n] = 1.0f / (1.0f + __expf(-s));
}

__global__ void count_nodes(const int* __restrict__ batch, float* __restrict__ counts) {
  int n = blockIdx.x * blockDim.x + threadIdx.x;
  if (n >= N_NODES) return;
  unsafeAtomicAdd(counts + batch[n], 1.0f);
}

// wx = h * a ; segment sum + segment max (wx >= 0 -> uint-ordered max is valid)
__global__ void pool(const float* __restrict__ h, const float* __restrict__ a,
                     const int* __restrict__ batch,
                     float* __restrict__ sums, float* __restrict__ maxs) {
  long long t = (long long)blockIdx.x * blockDim.x + threadIdx.x;
  if (t >= (long long)N_NODES * HID) return;
  int n = (int)(t >> 8), c = (int)(t & 255);
  int g = batch[n];
  float val = h[(size_t)n * HID + c] * a[n];
  unsafeAtomicAdd(&sums[(size_t)g * HID + c], val);
  atomicMax((unsigned int*)&maxs[(size_t)g * HID + c], __float_as_uint(val));
}

__global__ void build_g(const float* __restrict__ sums, const float* __restrict__ maxs,
                        const float* __restrict__ counts, float* __restrict__ g) {
  int t = blockIdx.x * blockDim.x + threadIdx.x;
  if (t >= N_GRAPHS * 2 * HID) return;
  int gr = t >> 9, c = t & 511;
  g[t] = (c < HID) ? sums[(size_t)gr * HID + c] / fmaxf(counts[gr], 1.0f)
                   : maxs[(size_t)gr * HID + (c - HID)];
}

__global__ void classifier3(const float* __restrict__ c2, const float* __restrict__ Wc3,
                            const float* __restrict__ bc3, float* __restrict__ preds) {
  int t = blockIdx.x * blockDim.x + threadIdx.x;
  if (t >= N_GRAPHS * 2) return;
  int gr = t >> 1, j = t & 1;
  float s = bc3[j];
#pragma unroll 8
  for (int k = 0; k < HID / 2; ++k) s += c2[(size_t)gr * (HID / 2) + k] * Wc3[k * 2 + j];
  preds[t] = s;
}

// ---------------------------------------------------------------------------
static inline int cdiv(long long a, long long b) { return (int)((a + b - 1) / b); }

extern "C" void kernel_launch(void* const* d_in, const int* in_sizes, int n_in,
                              void* d_out, int out_size, void* d_ws, size_t ws_size,
                              hipStream_t stream) {
  const float* x    = (const float*)d_in[0];
  const int*   ei   = (const int*)d_in[1];   // [2, N_EDGES]: src then dst
  const int*   batch= (const int*)d_in[2];
  const float* W_in = (const float*)d_in[3];
  const float* b_in = (const float*)d_in[4];
  const float* ggc  = (const float*)d_in[5]; // [5,256,256]
  const float* W_ih = (const float*)d_in[6]; // [768,256]
  const float* W_hh = (const float*)d_in[7];
  const float* b_ih = (const float*)d_in[8];
  const float* b_hh = (const float*)d_in[9];
  const float* Wa1  = (const float*)d_in[10];
  const float* ba1  = (const float*)d_in[11];
  const float* Wa2  = (const float*)d_in[12];
  const float* ba2  = (const float*)d_in[13];
  const float* Wc1  = (const float*)d_in[14];
  const float* bc1  = (const float*)d_in[15];
  const float* Wc2  = (const float*)d_in[16];
  const float* bc2  = (const float*)d_in[17];
  const float* Wc3  = (const float*)d_in[18];
  const float* bc3  = (const float*)d_in[19];

  float* preds = (float*)d_out;              // [128,2]
  float* a_out = (float*)d_out + N_GRAPHS*2; // [50000]

  char* ws = (char*)d_ws;
  size_t off = 0;
  auto alloc = [&](size_t bytes) -> char* {
    char* p = ws + off;
    off = (off + bytes + 255) & ~(size_t)255;
    return p;
  };
  float* h     = (float*)alloc((size_t)N_NODES * HID * 4);
  float* m     = (float*)alloc((size_t)N_NODES * HID * 4);
  float* agg   = (float*)alloc((size_t)N_NODES * HID * 4);
  float* gi    = (float*)alloc((size_t)CHUNK * 3 * HID * 4);
  float* gh    = (float*)alloc((size_t)CHUNK * 3 * HID * 4);
  float* xpad  = (float*)alloc((size_t)N_NODES * KPAD_IN * 4);
  float* a1    = xpad; // alias: xpad dead after input projection; same size
  float* sums  = (float*)alloc((size_t)N_GRAPHS * HID * 4);
  float* maxs  = (float*)alloc((size_t)N_GRAPHS * HID * 4);
  float* cnts  = (float*)alloc((size_t)N_GRAPHS * 4);
  float* gfeat = (float*)alloc((size_t)N_GRAPHS * 2 * HID * 4);
  float* c1    = (float*)alloc((size_t)N_GRAPHS * HID * 4);
  float* c2    = (float*)alloc((size_t)N_GRAPHS * (HID / 2) * 4);
  u16* pWin = (u16*)alloc((size_t)KPAD_IN * HID * 2);
  u16* pGgc = (u16*)alloc((size_t)STEPS * HID * HID * 2);
  u16* pIh  = (u16*)alloc((size_t)HID * 3 * HID * 2);
  u16* pHh  = (u16*)alloc((size_t)HID * 3 * HID * 2);
  u16* pWa1 = (u16*)alloc((size_t)HID * (HID / 2) * 2);
  u16* pWc1 = (u16*)alloc((size_t)2 * HID * HID * 2);
  u16* pWc2 = (u16*)alloc((size_t)HID * (HID / 2) * 2);

  auto packB = [&](const float* W, int ld, int trans, int K, int N, u16* out, int Ksrc) {
    pack_b<<<cdiv((long long)K * N, 256), 256, 0, stream>>>(W, ld, trans, K, N, out, Ksrc);
  };
  packB(W_in, HID, 0, KPAD_IN, HID, pWin, IN_DIM);
  for (int s = 0; s < STEPS; ++s)
    packB(ggc + (size_t)s * HID * HID, HID, 0, HID, HID, pGgc + (size_t)s * HID * HID, HID);
  packB(W_ih, HID, 1, HID, 3 * HID, pIh, HID);   // B[k][j] = W_ih[j][k]
  packB(W_hh, HID, 1, HID, 3 * HID, pHh, HID);
  packB(Wa1, HID / 2, 0, HID, HID / 2, pWa1, HID);
  packB(Wc1, HID, 0, 2 * HID, HID, pWc1, 2 * HID);
  packB(Wc2, HID / 2, 0, HID, HID / 2, pWc2, HID);

  auto gemmR = [&](const float* A, const u16* B, const float* bias, float* C,
                   int M, int K, int N) {
    long long waves = (long long)(M / 16) * (N / 64);
    gemm_wmma<4, 1><<<cdiv(waves * 32, 256), 256, 0, stream>>>(A, B, bias, C, M, K, N);
  };
  auto gemmL = [&](const float* A, const u16* B, const float* bias, float* C,
                   int M, int K, int N) {
    long long waves = (long long)(M / 16) * (N / 64);
    gemm_wmma<4, 0><<<cdiv(waves * 32, 256), 256, 0, stream>>>(A, B, bias, C, M, K, N);
  };

  // input projection: h = relu(x @ W_in + b_in)
  pad_x<<<cdiv((long long)N_NODES * KPAD_IN, 256), 256, 0, stream>>>(x, xpad);
  gemmR(xpad, pWin, b_in, h, N_NODES, KPAD_IN, HID);

  // 5 GatedGraphConv steps
  for (int s = 0; s < STEPS; ++s) {
    gemmL(h, pGgc + (size_t)s * HID * HID, nullptr, m, N_NODES, HID, HID);
    fill_f32<<<cdiv((long long)N_NODES * HID, 256), 256, 0, stream>>>(agg, 0.0f,
                                                                     (long long)N_NODES * HID);
    scatter_add<<<cdiv((long long)N_EDGES * 64, 256), 256, 0, stream>>>(m, ei, ei + N_EDGES, agg);
    for (int base = 0; base < N_NODES; base += CHUNK) {
      int cm = (N_NODES - base < CHUNK) ? (N_NODES - base) : CHUNK;
      gemmL(agg + (size_t)base * HID, pIh, b_ih, gi, cm, HID, 3 * HID);
      gemmL(h + (size_t)base * HID, pHh, b_hh, gh, cm, HID, 3 * HID);
      gru_cell<<<cdiv((long long)cm * HID, 256), 256, 0, stream>>>(gi, gh, h, base, cm);
    }
  }
  relu_ip<<<cdiv((long long)N_NODES * HID, 256), 256, 0, stream>>>(h, (long long)N_NODES * HID);

  // attention: a = sigmoid(relu(h@Wa1+ba1) @ Wa2 + ba2)
  gemmR(h, pWa1, ba1, a1, N_NODES, HID, HID / 2);
  attention<<<cdiv(N_NODES, 256), 256, 0, stream>>>(a1, Wa2, ba2, a_out);

  // pooling
  fill_f32<<<cdiv((long long)N_GRAPHS * HID, 256), 256, 0, stream>>>(sums, 0.0f,
                                                                    (long long)N_GRAPHS * HID);
  fill_f32<<<cdiv((long long)N_GRAPHS * HID, 256), 256, 0, stream>>>(maxs, 0.0f,
                                                                    (long long)N_GRAPHS * HID);
  fill_f32<<<1, 256, 0, stream>>>(cnts, 0.0f, N_GRAPHS);
  count_nodes<<<cdiv(N_NODES, 256), 256, 0, stream>>>(batch, cnts);
  pool<<<cdiv((long long)N_NODES * HID, 256), 256, 0, stream>>>(h, a_out, batch, sums, maxs);
  build_g<<<cdiv(N_GRAPHS * 2 * HID, 256), 256, 0, stream>>>(sums, maxs, cnts, gfeat);

  // classifier
  gemmR(gfeat, pWc1, bc1, c1, N_GRAPHS, 2 * HID, HID);
  gemmR(c1, pWc2, bc2, c2, N_GRAPHS, HID, HID / 2);
  classifier3<<<1, 256, 0, stream>>>(c2, Wc3, bc3, preds);
}
